// VQ_single_10634339025166
// MI455X (gfx1250) — matrix operand order repaired and checked by reading, not compiled, not run on hardware
//
#include <hip/hip_runtime.h>
#include <stddef.h>

// ---------------------------------------------------------------------------
// VQ-VAE forward for MI455X (gfx1250, wave32, WMMA + TDM).
// All heavy contractions -> V_WMMA_F32_16X16X32_BF16 implicit GEMMs.
// ---------------------------------------------------------------------------

typedef __bf16 bf16;
typedef unsigned long long u64;
typedef __attribute__((ext_vector_type(16))) __bf16 v16bf;
typedef __attribute__((ext_vector_type(8)))  float  v8f;
typedef __attribute__((ext_vector_type(4)))  unsigned int v4u;
typedef __attribute__((ext_vector_type(8)))  unsigned int v8u;

// ---------------------------------------------------------------------------
// Weight repack: fp32 OIDHW -> bf16 "fragment ready" B operand.
// Layout: Bw[parity][kblk][nblk][lane(32)][16 bf16]  (32B contiguous per lane)
// B fragment as packed here (contiguous k order, same as A): lanes 0-15 cover K=0..15 (n = lane&15),
// lanes 16-31 cover K=16..31; VGPR j holds K = kh*16 + 2j, 2j+1.
// RM: 0 = conv weight (taps 64), 1 = deconv per-parity (taps 8), 2 = codebook.
// ---------------------------------------------------------------------------
template<int RM>
__global__ void repack_k(const float* __restrict__ src, bf16* __restrict__ dst,
                         int CIN, int NB, int KBLK)
{
    const int l  = threadIdx.x;          // lane 0..31
    const int kb = blockIdx.x;
    const int nb = blockIdx.y;
    const int par = blockIdx.z;
    const int n  = nb * 16 + (l & 15);
    const int kh = l >> 4;

    const size_t base = ((size_t)((size_t)par * KBLK + kb) * NB + nb) * 512 + (size_t)l * 16;

#pragma unroll
    for (int j = 0; j < 8; ++j) {
#pragma unroll
        for (int d2 = 0; d2 < 2; ++d2) {
            const int k = kb * 32 + kh * 16 + 2 * j + d2;
            long si;
            if (RM == 0) {                       // conv: K = t*CIN + ci, w[o][i][t]
                const int t = k / CIN, ci = k % CIN;
                si = ((long)n * CIN + ci) * 64 + t;
            } else if (RM == 1) {                // deconv parity class
                const int t = k / CIN, ci = k % CIN;
                const int sz = t >> 2, sy = (t >> 1) & 1, sx = t & 1;
                const int pz = (par >> 2) & 1, py = (par >> 1) & 1, px = par & 1;
                const int tt = (pz + 2 * sz) * 16 + (py + 2 * sy) * 4 + (px + 2 * sx);
                si = ((long)n * CIN + ci) * 64 + tt;
            } else {                             // codebook[n][k] -> B[k][n]
                si = (long)n * 256 + k;
            }
            dst[base + 2 * j + d2] = (bf16)src[si];
        }
    }
}

// ---------------------------------------------------------------------------
// conv1: Cin=1, k=4, s=2, p=1, 1 -> 128 channels, bias + relu, bf16 out
// ---------------------------------------------------------------------------
__global__ __launch_bounds__(256) void conv1_k(const float* __restrict__ x,
                                               const float* __restrict__ w,
                                               const float* __restrict__ bias,
                                               bf16* __restrict__ out)
{
    const int tid = threadIdx.x;
    const int co  = tid & 127;
    const int vl  = tid >> 7;
    const int rg  = blockIdx.x * 2 + vl;
    const int ox  = rg & 31, oy = (rg >> 5) & 31, oz = (rg >> 10) & 31, b = rg >> 15;

    float acc = 0.f;
#pragma unroll 4
    for (int t = 0; t < 64; ++t) {
        const int kz = t >> 4, ky = (t >> 2) & 3, kx = t & 3;
        const int iz = 2 * oz + kz - 1, iy = 2 * oy + ky - 1, ix = 2 * ox + kx - 1;
        float xv = 0.f;
        if ((unsigned)iz < 64u && (unsigned)iy < 64u && (unsigned)ix < 64u)
            xv = x[(((long)b * 64 + iz) * 64 + iy) * 64 + ix];
        acc += xv * w[co * 64 + t];
    }
    acc += bias[co];
    out[(size_t)rg * 128 + co] = (bf16)fmaxf(acc, 0.f);
}

// ---------------------------------------------------------------------------
// Implicit-GEMM WMMA kernel. Block = 256 threads = 8 waves (2M x 4N wave grid),
// each wave computes a 16(M) x 32(N) C tile (2 WMMA per K-step) -> block tile
// is 32(M) x 128(N). K-loop step 32 (one tap, 32 channels).
// MODE: 0=conv2  1=conv3  2=VQ-dist(TDM A-stage)  3=deconv1  4=deconv2
//       5=deconv3 (fused 1x1 output conv via atomics)
// ---------------------------------------------------------------------------
template<int MODE>
__global__ __launch_bounds__(256) void gemm_k(
    const bf16*  __restrict__ A,    const bf16*  __restrict__ Bw,
    const float* __restrict__ bias, bf16*        __restrict__ obf,
    float*       __restrict__ of32, const float* __restrict__ wo,
    float*       __restrict__ oat)
{
    constexpr bool ISCONV   = (MODE == 0 || MODE == 1);
    constexpr bool ISDECONV = (MODE >= 3);
    constexpr bool ISDIST   = (MODE == 2);
    constexpr int  CIN   = (MODE == 0) ? 128 : (MODE == 5) ? 128 : 256;
    constexpr int  DIN   = (MODE == 0) ? 32 : (MODE == 1) ? 16 :
                           (MODE == 3) ? 8  : (MODE == 4) ? 16 :
                           (MODE == 5) ? 32 : 1;
    constexpr int  COUT  = (MODE == 4 || MODE == 5) ? 128 : (MODE == 2) ? 1024 : 256;
    constexpr int  TAPS  = ISCONV ? 64 : (ISDECONV ? 8 : 1);
    constexpr int  KTOT  = ISDIST ? 256 : CIN * TAPS;
    constexpr int  KSTEPS = KTOT / 32;
    constexpr int  CPT   = ISDIST ? 8 : (CIN / 32);
    constexpr int  NB    = COUT / 16;
    constexpr bool RELU  = (MODE == 0 || MODE == 3 || MODE == 4);

    __shared__ __align__(16) bf16 As[32][40];   // 32 rows x 32 K, 80B row stride

    const int tid  = threadIdx.x;
    const int lane = tid & 31;
    const int wv   = tid >> 5;
    const int wm   = wv >> 2, wn = wv & 3;

    // staging: each thread fills one 8-byte chunk of one LDS row per K-step
    const int sr = tid >> 3, spart = tid & 7;
    const int srg = blockIdx.x * 32 + sr;
    int sb = 0, s0 = 0, s1 = 0, s2 = 0;
    if constexpr (ISCONV) {
        constexpr int DO = DIN / 2;
        s2 = srg % DO; s1 = (srg / DO) % DO;
        s0 = (srg / (DO * DO)) % DO; sb = srg / (DO * DO * DO);
    } else if constexpr (ISDECONV) {
        s2 = srg % DIN; s1 = (srg / DIN) % DIN;
        s0 = (srg / (DIN * DIN)) % DIN; sb = srg / (DIN * DIN * DIN);
    }
    int pz = 0, py = 0, px = 0;
    if constexpr (ISDECONV) {
        pz = (blockIdx.z >> 2) & 1; py = (blockIdx.z >> 1) & 1; px = blockIdx.z & 1;
    }

    v8f acc0 = {}, acc1 = {};
    const int nbG = blockIdx.y * 8 + wn * 2;    // two adjacent 16-col B blocks

    for (int ks = 0; ks < KSTEPS; ++ks) {
        // ---- stage A tile ----
        if constexpr (ISDIST) {
            // Regular 2D tile: Tensor Data Mover, issued by wave 0 only.
            if (wv == 0) {
                const u64 gaddr = (u64)(uintptr_t)A +
                                  ((u64)blockIdx.x * 32 * 256 + (u64)ks * 32) * 2;
                const unsigned ldsa = (unsigned)(uintptr_t)&As[0][0];
                v4u g0;
                g0.x = 1u;                                      // count=1, user D#
                g0.y = ldsa;                                    // lds_addr (bytes)
                g0.z = (unsigned)gaddr;                         // global_addr[31:0]
                g0.w = ((unsigned)(gaddr >> 32) & 0x01FFFFFFu)  // global_addr[56:32]
                       | 0x80000000u;                           // type=2 (image)
                v8u g1;
                g1[0] = (1u << 16)      // data_size = 2 bytes
                      | (1u << 20)      // pad_enable
                      | (3u << 22)      // pad_interval: 16 dwords (64B row)
                      | (3u << 25);     // pad_amount: 4 dwords (16B -> 80B stride)
                g1[1] = 32u << 16;      // tensor_dim0 = 32 elems
                g1[2] = 32u << 16;      // tensor_dim1 = 32 rows
                g1[3] = 32u << 16;      // tile_dim0 = 32 elems
                g1[4] = 32u;            // tile_dim1 = 32, tile_dim2 = 0
                g1[5] = 256u;           // tensor_dim0_stride = 256 elems
                g1[6] = 0u;
                g1[7] = 0u;
                v4u gz = {0u, 0u, 0u, 0u};
                asm volatile("tensor_load_to_lds %0, %1, %2, %3"
                             :: "s"(g0), "s"(g1), "s"(gz), "s"(gz)
                             : "memory");
                __builtin_amdgcn_s_wait_tensorcnt(0);
            }
        } else {
            const int t = ks / CPT, ci0 = (ks % CPT) * 32;
            long base = 0; bool ok = true;
            if constexpr (ISCONV) {
                const int kz = t >> 4, ky = (t >> 2) & 3, kx = t & 3;
                const int iz = 2 * s0 + kz - 1, iy = 2 * s1 + ky - 1, ix = 2 * s2 + kx - 1;
                ok = (unsigned)iz < (unsigned)DIN && (unsigned)iy < (unsigned)DIN &&
                     (unsigned)ix < (unsigned)DIN;
                base = ((((long)sb * DIN + iz) * DIN + iy) * DIN + ix) * CIN + ci0;
            } else {
                const int sz = t >> 2, sy = (t >> 1) & 1, sx = t & 1;
                const int iz = s0 + sz - 1 + pz, iy = s1 + sy - 1 + py, ix = s2 + sx - 1 + px;
                ok = (unsigned)iz < (unsigned)DIN && (unsigned)iy < (unsigned)DIN &&
                     (unsigned)ix < (unsigned)DIN;
                base = ((((long)sb * DIN + iz) * DIN + iy) * DIN + ix) * CIN + ci0;
            }
            u64 chunk = 0;
            if (ok) chunk = *(const u64*)(A + base + spart * 4);
            *(u64*)&As[sr][spart * 4] = chunk;     // single b64 LDS store
        }
        __syncthreads();

        // ---- build fragments ----
        const int m = lane & 15, kh = lane >> 4;
        const bf16* arow = &As[wm * 16 + m][0];
        v16bf afrag, bfrag0, bfrag1;
#pragma unroll
        for (int j = 0; j < 8; ++j) {
            afrag[j]     = arow[kh * 16 + j];         // K = kh*16 + j
            afrag[8 + j] = arow[kh * 16 + 8 + j];     // K = kh*16 + 8 + j
        }
        {
            const size_t boff =
                (((size_t)blockIdx.z * KSTEPS + ks) * NB + nbG) * 512 + (size_t)lane * 16;
            bfrag0 = *(const v16bf*)(Bw + boff);
            bfrag1 = *(const v16bf*)(Bw + boff + 512);
            if (ks + 1 < KSTEPS)
                __builtin_prefetch(Bw + boff + (size_t)NB * 512, 0, 1);
        }
        acc0 = __builtin_amdgcn_wmma_f32_16x16x32_bf16(
                   false, afrag, false, bfrag0, (short)0, acc0, false, false);
        acc1 = __builtin_amdgcn_wmma_f32_16x16x32_bf16(
                   false, afrag, false, bfrag1, (short)0, acc1, false, false);
        __syncthreads();
    }

    // ---- epilogue: C 16x16 f32, lane -> (row = (lane>>4)*8 + v, col = lane&15) ----
    const int col = lane & 15;
    const int rb  = (lane >> 4) * 8;
    const int co0 = blockIdx.y * 128 + wn * 32 + col;
    const float bs0 = bias ? bias[co0] : 0.f;
    const float bs1 = bias ? bias[co0 + 16] : 0.f;

#pragma unroll
    for (int v = 0; v < 8; ++v) {
        const int rowT = wm * 16 + rb + v;
        const int rg   = blockIdx.x * 32 + rowT;
        float v0 = acc0[v] + bs0;
        float v1 = acc1[v] + bs1;
        if constexpr (RELU) { v0 = fmaxf(v0, 0.f); v1 = fmaxf(v1, 0.f); }

        if constexpr (ISDIST) {
            of32[(size_t)rg * COUT + co0]      = acc0[v];        // raw dot products
            of32[(size_t)rg * COUT + co0 + 16] = acc1[v];
        } else if constexpr (ISCONV) {
            obf[(size_t)rg * COUT + co0]      = (bf16)v0;
            obf[(size_t)rg * COUT + co0 + 16] = (bf16)v1;
            if constexpr (MODE == 1) {
                of32[(size_t)rg * COUT + co0]      = v0;         // z fp32 copy
                of32[(size_t)rg * COUT + co0 + 16] = v1;
            }
        } else {
            const int mx = rg % DIN, my = (rg / DIN) % DIN;
            const int mz = (rg / (DIN * DIN)) % DIN, b = rg / (DIN * DIN * DIN);
            const int z = 2 * mz + pz, y = 2 * my + py, x = 2 * mx + px;
            constexpr int DF = 2 * DIN;
            const size_t vox = (((size_t)b * DF + z) * DF + y) * DF + x;
            if constexpr (MODE == 5) {
                atomicAdd(&oat[vox], v0 * wo[co0] + v1 * wo[co0 + 16]);  // fused 1x1
            } else {
                obf[vox * COUT + co0]      = (bf16)v0;
                obf[vox * COUT + co0 + 16] = (bf16)v1;
            }
        }
    }
}

// ---------------------------------------------------------------------------
// codebook row norms
// ---------------------------------------------------------------------------
__global__ __launch_bounds__(256) void cbnorm_k(const float* __restrict__ cb,
                                                float* __restrict__ out)
{
    const int c = blockIdx.x * 256 + threadIdx.x;
    float s = 0.f;
    for (int d = 0; d < 256; ++d) { const float v = cb[(size_t)c * 256 + d]; s += v * v; }
    out[c] = s;
}

// ---------------------------------------------------------------------------
// VQ: argmin over codes, gather q (bf16 for decoder), accumulate (q-z)^2
// ---------------------------------------------------------------------------
__global__ __launch_bounds__(256) void vq_k(const float* __restrict__ dots,
                                            const float* __restrict__ cbnorm,
                                            const float* __restrict__ cb,
                                            const float* __restrict__ zf,
                                            bf16* __restrict__ qbf,
                                            float* __restrict__ lossAcc)
{
    __shared__ float sv[256];
    __shared__ int   si[256];
    const int r = blockIdx.x, tid = threadIdx.x;

    float best = 3.4e38f; int bi = 0;
    for (int c = tid; c < 1024; c += 256) {
        const float s = cbnorm[c] - 2.0f * dots[(size_t)r * 1024 + c];
        if (s < best) { best = s; bi = c; }
    }
    sv[tid] = best; si[tid] = bi; __syncthreads();
    for (int s = 128; s > 0; s >>= 1) {
        if (tid < s) {
            if (sv[tid + s] < sv[tid] ||
                (sv[tid + s] == sv[tid] && si[tid + s] < si[tid])) {
                sv[tid] = sv[tid + s]; si[tid] = si[tid + s];
            }
        }
        __syncthreads();
    }
    const int idx = si[0];
    __syncthreads();

    const float q = cb[(size_t)idx * 256 + tid];
    qbf[(size_t)r * 256 + tid] = (bf16)q;
    const float diff = q - zf[(size_t)r * 256 + tid];
    sv[tid] = diff * diff; __syncthreads();
    for (int s = 128; s > 0; s >>= 1) {
        if (tid < s) sv[tid] += sv[tid + s];
        __syncthreads();
    }
    if (tid == 0) atomicAdd(lossAcc, sv[0]);
}

// ---------------------------------------------------------------------------
__global__ __launch_bounds__(256) void init_k(float* __restrict__ out,
                                              const float* __restrict__ bo,
                                              float* __restrict__ lossAcc, int n)
{
    const int i = blockIdx.x * 256 + threadIdx.x;
    if (i == 0) { lossAcc[0] = 0.f; out[0] = 0.f; }
    if (i < n) out[1 + i] = bo[0];
}

__global__ void fin_k(float* __restrict__ out, const float* __restrict__ lossAcc)
{
    // eq_loss = q_latent + CC*e_latent ; both equal mean((q-z)^2) in forward
    out[0] = 1.25f * lossAcc[0] / 262144.0f;   // / (1024*256)
}

// ---------------------------------------------------------------------------
extern "C" void kernel_launch(void* const* d_in, const int* in_sizes, int n_in,
                              void* d_out, int out_size, void* d_ws, size_t ws_size,
                              hipStream_t stream)
{
    (void)in_sizes; (void)n_in; (void)out_size;

    const float* x   = (const float*)d_in[0];
    const float* cb  = (const float*)d_in[1];
    const float* w1  = (const float*)d_in[2];
    const float* b1  = (const float*)d_in[3];
    const float* w2  = (const float*)d_in[4];
    const float* b2  = (const float*)d_in[5];
    const float* w3  = (const float*)d_in[6];
    const float* b3  = (const float*)d_in[7];
    const float* dw1 = (const float*)d_in[8];
    const float* db1 = (const float*)d_in[9];
    const float* dw2 = (const float*)d_in[10];
    const float* db2 = (const float*)d_in[11];
    const float* dw3 = (const float*)d_in[12];
    const float* db3 = (const float*)d_in[13];
    const float* wo  = (const float*)d_in[14];
    const float* bo  = (const float*)d_in[15];
    float* out = (float*)d_out;

    char* ws = (char*)d_ws;
    size_t o = 0;
    auto alloc = [&](size_t bytes) -> char* {
        char* p = ws + o;
        o = (o + bytes + 255) & ~(size_t)255;
        return p;
    };

    bf16*  act1  = (bf16*) alloc(8388608ull * 2);   // [2*32^3,128]
    bf16*  act2  = (bf16*) alloc(2097152ull * 2);   // [2*16^3,256]
    bf16*  zbf   = (bf16*) alloc(262144ull  * 2);   // [1024,256]
    float* zf32  = (float*)alloc(262144ull  * 4);
    float* dots  = (float*)alloc(1048576ull * 4);   // [1024,1024]
    bf16*  qbf   = (bf16*) alloc(262144ull  * 2);
    bf16*  d1a   = (bf16*) alloc(2097152ull * 2);   // [2*16^3,256]
    bf16*  d2a   = (bf16*) alloc(8388608ull * 2);   // [2*32^3,128]
    bf16*  BwC2  = (bf16*) alloc(2097152ull * 2);   // K=8192,N=256
    bf16*  BwC3  = (bf16*) alloc(4194304ull * 2);   // K=16384,N=256
    bf16*  BwD1  = (bf16*) alloc(4194304ull * 2);   // 8 x (K=2048,N=256)
    bf16*  BwD2  = (bf16*) alloc(2097152ull * 2);   // 8 x (K=2048,N=128)
    bf16*  BwD3  = (bf16*) alloc(1048576ull * 2);   // 8 x (K=1024,N=128)
    bf16*  BwCB  = (bf16*) alloc(262144ull  * 2);   // K=256,N=1024
    float* cbn   = (float*)alloc(1024ull    * 4);
    float* lacc  = (float*)alloc(64ull      * 4);
    if (o > ws_size) return;   // workspace too small: do nothing (deterministic)

    // --- init output (recon preset to bias; deconv3 epilogue atomically adds) ---
    init_k<<<dim3(2048), dim3(256), 0, stream>>>(out, bo, lacc, 524288);

    // --- weight / codebook repack into fragment-ready bf16 ---
    repack_k<0><<<dim3(256, 16, 1), dim3(32), 0, stream>>>(w2,  BwC2, 128, 16, 256);
    repack_k<0><<<dim3(512, 16, 1), dim3(32), 0, stream>>>(w3,  BwC3, 256, 16, 512);
    repack_k<1><<<dim3(64, 16, 8),  dim3(32), 0, stream>>>(dw1, BwD1, 256, 16, 64);
    repack_k<1><<<dim3(64, 8, 8),   dim3(32), 0, stream>>>(dw2, BwD2, 256, 8, 64);
    repack_k<1><<<dim3(32, 8, 8),   dim3(32), 0, stream>>>(dw3, BwD3, 128, 8, 32);
    repack_k<2><<<dim3(8, 64, 1),   dim3(32), 0, stream>>>(cb,  BwCB, 256, 64, 8);
    cbnorm_k<<<dim3(4), dim3(256), 0, stream>>>(cb, cbn);

    // --- encoder ---
    conv1_k<<<dim3(32768), dim3(256), 0, stream>>>(x, w1, b1, act1);
    gemm_k<0><<<dim3(256, 2, 1), dim3(256), 0, stream>>>(act1, BwC2, b2, act2,
                                                         nullptr, nullptr, nullptr);
    gemm_k<1><<<dim3(32, 2, 1), dim3(256), 0, stream>>>(act2, BwC3, b3, zbf,
                                                        zf32, nullptr, nullptr);

    // --- VQ: distances (WMMA GEMM, TDM A-stage), argmin + gather + loss ---
    gemm_k<2><<<dim3(32, 8, 1), dim3(256), 0, stream>>>(zbf, BwCB, nullptr, nullptr,
                                                        dots, nullptr, nullptr);
    vq_k<<<dim3(1024), dim3(256), 0, stream>>>(dots, cbn, cb, zf32, qbf, lacc);

    // --- decoder (parity-decomposed transpose convs) ---
    gemm_k<3><<<dim3(32, 2, 8),   dim3(256), 0, stream>>>(qbf, BwD1, db1, d1a,
                                                          nullptr, nullptr, nullptr);
    gemm_k<4><<<dim3(256, 1, 8),  dim3(256), 0, stream>>>(d1a, BwD2, db2, d2a,
                                                          nullptr, nullptr, nullptr);
    gemm_k<5><<<dim3(2048, 1, 8), dim3(256), 0, stream>>>(d2a, BwD3, db3, nullptr,
                                                          nullptr, wo, out + 1);

    // --- loss finalize ---
    fin_k<<<dim3(1), dim3(1), 0, stream>>>(out, lacc);
}
